// RLDocumentRetriever_14542759264482
// MI455X (gfx1250) — compile-verified, matrix operand order checked by reading, not run on hardware
//
#include <hip/hip_runtime.h>
#include <hip/hip_bf16.h>
#include <math.h>

typedef _Float16 v16h __attribute__((ext_vector_type(16)));
typedef _Float16 v8h  __attribute__((ext_vector_type(8)));
typedef float    v8f  __attribute__((ext_vector_type(8)));
typedef int      v4i  __attribute__((ext_vector_type(4)));

#define V_SZ   50000
#define D_SZ   300
#define H_SZ   128
#define KPAD   608      // 600 padded to 19*32
#define NKC    19
#define NCAND  1000
#define NCPAD  1008
#define NFB    5120     // 10*512 feedback tokens

#define GAS __attribute__((address_space(1)))
#define LAS __attribute__((address_space(3)))

#if defined(__gfx1250__) && __has_builtin(__builtin_amdgcn_global_load_async_to_lds_b128)
#define HAVE_ASYNC_LDS 1
#else
#define HAVE_ASYNC_LDS 0
#endif

// ---------------- workspace layout (bytes, 256-aligned) ----------------
#define OFF_WEIGHTS  0u         // 50000 f32
#define OFF_QCTX     200704u    // 300 f32
#define OFF_TOPIDS   203008u    // 10 i32
#define OFF_TOPW     203264u    // 10 f32
#define OFF_VALUE    203520u    // 1 f32
#define OFF_EQF      203776u    // 300 f32
#define OFF_EQH      205056u    // 300 f16
#define OFF_TSW1P    205824u    // packed W1 (ts): 8*19*32*16 f16 = 155648 B
#define OFF_DSW1P    361472u    // packed W1 (ds): 155648 B
#define OFF_DSA      517120u    // 1008*608 f16 = 1225728 B
#define OFF_SCORES   1742848u   // 1008 f32

// ---------------- tiny utility kernels ----------------
__global__ void k_zero_weights(float* w) {
    int i = blockIdx.x * 256 + threadIdx.x;
    if (i < V_SZ) w[i] = 0.f;
}

// Pack W1 [600,128]f32 into WMMA-B fragment-native f16 layout:
//   [ntile(8)][kc(19)][lane(32)][e(16)]
//   k = kc*32 + (e>=8 ? 16 : 0) + (lane>=16 ? 8 : 0) + (e&7)
//   col = ntile*16 + (lane&15)
__global__ void k_pack_w1(const float* __restrict__ tsw1,
                          const float* __restrict__ dsw1,
                          _Float16* tsp, _Float16* dsp) {
    int i = blockIdx.x * 256 + threadIdx.x;
    if (i >= 8 * NKC * 32 * 16) return;
    int e    = i & 15;
    int lane = (i >> 4) & 31;
    int t    = i >> 9;                 // nt*NKC + kc
    int kc   = t % NKC, nt = t / NKC;
    int k    = kc * 32 + ((e & 8) ? 16 : 0) + ((lane & 16) ? 8 : 0) + (e & 7);
    int col  = nt * 16 + (lane & 15);
    float a = (k < 600) ? tsw1[k * H_SZ + col] : 0.f;
    float b = (k < 600) ? dsw1[k * H_SZ + col] : 0.f;
    tsp[i] = (_Float16)a;
    dsp[i] = (_Float16)b;
}

__global__ __launch_bounds__(320)
void k_qctx(const int* __restrict__ query, const float* __restrict__ emb,
            float* qctx) {
    __shared__ int q[64];
    int tid = threadIdx.x;
    if (tid < 64) q[tid] = query[tid];
    __syncthreads();
    if (tid < D_SZ) {
        float acc = 0.f;
        for (int i = 0; i < 64; ++i) acc += emb[(long)q[i] * D_SZ + tid];
        qctx[tid] = acc * (1.f / 64.f);
    }
}

// ---------------- WMMA fragment helpers ----------------
// 16-bit A 16x32 lane layout (ISA 7.12.2):
//   lanes 0-15 : M=lane,    VGPR0-3 K=0..7,  VGPR4-7 K=16..23
//   lanes16-31 : M=lane-16, VGPR0-3 K=8..15, VGPR4-7 K=24..31
__device__ inline v16h frag_a_from_lds(const _Float16* At, int lane, int kc) {
    int m  = lane & 15;
    int kb = kc * 32 + ((lane & 16) ? 8 : 0);
    const _Float16* p = At + m * KPAD + kb;   // 16B-aligned
    v8h lo = *(const v8h*)(p);
    v8h hi = *(const v8h*)(p + 16);
    v16h a;
#pragma unroll
    for (int e = 0; e < 8; ++e) { a[e] = lo[e]; a[8 + e] = hi[e]; }
    return a;
}

// B fragment: contiguous 32B per lane from the packed layout
__device__ inline v16h frag_b_packed(const _Float16* __restrict__ w1p,
                                     int nt, int kc, int lane) {
    const _Float16* p = w1p + (((nt * NKC) + kc) * 32 + lane) * 16;
    v8h lo = *(const v8h*)(p);
    v8h hi = *(const v8h*)(p + 8);
    v16h b;
#pragma unroll
    for (int e = 0; e < 8; ++e) { b[e] = lo[e]; b[8 + e] = hi[e]; }
    return b;
}

// ---------------- term scorer: [5120,608]f16 x [608,128]f16 via WMMA ----------------
__global__ __launch_bounds__(128)
void k_term_score(const int* __restrict__ fb, const float* __restrict__ emb,
                  const float* __restrict__ qctx,
                  const _Float16* __restrict__ w1p, const float* __restrict__ b1,
                  const float* __restrict__ w2, const float* __restrict__ b2v,
                  float* weights) {
    __shared__ __align__(16) _Float16 At[16 * KPAD];
    __shared__ float Hid[16 * H_SZ];
    __shared__ float Red[128];
    __shared__ int   tok[16];

    const int tid = threadIdx.x, lane = tid & 31, wave = tid >> 5;
    const int rowBase = blockIdx.x * 16;

    if (tid < 16) tok[tid] = fb[rowBase + tid];
    __syncthreads();

    // stage A tile: [emb(tok) | query_ctx | 0-pad] in f16 (gather + convert)
    for (int i = tid; i < 16 * KPAD; i += 128) {
        int r = i / KPAD, k = i - r * KPAD;
        float v;
        if (k < D_SZ)       v = emb[(long)tok[r] * D_SZ + k];
        else if (k < 600)   v = qctx[k - D_SZ];
        else                v = 0.f;
        At[i] = (_Float16)v;
    }
    __syncthreads();

    for (int nt = wave; nt < 8; nt += 4) {
        int col = nt * 16 + (lane & 15);
        float bias = b1[col];
        v8f c;
#pragma unroll
        for (int r = 0; r < 8; ++r) c[r] = bias;
#pragma unroll
        for (int kc = 0; kc < NKC; ++kc) {
            v16h a = frag_a_from_lds(At, lane, kc);
            v16h b = frag_b_packed(w1p, nt, kc, lane);
            c = __builtin_amdgcn_wmma_f32_16x16x32_f16(
                    false, a, false, b, (short)0, c, false, false);
        }
        int mb = (lane & 16) ? 8 : 0;
#pragma unroll
        for (int r = 0; r < 8; ++r) {
            float h = c[r];
            Hid[(mb + r) * H_SZ + col] = h > 0.f ? h : 0.f;
        }
    }
    __syncthreads();

    {
        int r = tid >> 3, part = tid & 7;
        float acc = 0.f;
#pragma unroll
        for (int h = 0; h < 16; ++h)
            acc += Hid[r * H_SZ + part * 16 + h] * w2[part * 16 + h];
        Red[tid] = acc;
    }
    __syncthreads();
    if (tid < 16) {
        float s = b2v[0];
#pragma unroll
        for (int p = 0; p < 8; ++p) s += Red[tid * 8 + p];
        s = 1.0f / (1.0f + __expf(-s));
        atomicAdd(&weights[tok[tid]], s);
    }
}

// ---------------- top-k (k=10) over 50000 weights ----------------
__global__ __launch_bounds__(256)
void k_topk(float* weights, int* topIds, float* topW) {
    __shared__ float sv[256];
    __shared__ int   si[256];
    int tid = threadIdx.x;
    for (int j = 0; j < 10; ++j) {
        float bv = -1.f;               // weights >= 0 (sigmoid sums)
        int   bi = 0x7fffffff;
        for (int i = tid; i < V_SZ; i += 256) {
            float w = weights[i];
            if (w > bv || (w == bv && i < bi)) { bv = w; bi = i; }
        }
        sv[tid] = bv; si[tid] = bi;
        __syncthreads();
        for (int s = 128; s > 0; s >>= 1) {
            if (tid < s) {
                if (sv[tid + s] > sv[tid] ||
                    (sv[tid + s] == sv[tid] && si[tid + s] < si[tid])) {
                    sv[tid] = sv[tid + s]; si[tid] = si[tid + s];
                }
            }
            __syncthreads();
        }
        if (tid == 0) {
            topIds[j] = si[0];
            topW[j]   = sv[0];
            weights[si[0]] = -1e30f;
        }
        __syncthreads();
    }
}

// ---------------- expansion generator + value net ----------------
__global__ __launch_bounds__(320)
void k_expand(const int* __restrict__ topIds, const float* __restrict__ emb,
              const float* __restrict__ qctx,
              const float* __restrict__ egw1, const float* __restrict__ egb1,
              const float* __restrict__ egw2, const float* __restrict__ egb2,
              const float* __restrict__ vnw1, const float* __restrict__ vnb1,
              const float* __restrict__ vnw2, const float* __restrict__ vnb2,
              float* eqf, _Float16* eqh, float* valueOut) {
    __shared__ float egin[900];
    __shared__ float hid[128];
    __shared__ float eql[300];
    __shared__ int   ids[10];
    int tid = threadIdx.x;
    if (tid < 10) ids[tid] = topIds[tid];
    __syncthreads();
    if (tid < D_SZ) {
        float acc = 0.f;
        for (int j = 0; j < 10; ++j) acc += emb[(long)ids[j] * D_SZ + tid];
        float em = acc * 0.1f, qc = qctx[tid];
        egin[tid] = qc; egin[300 + tid] = em; egin[600 + tid] = qc * em;
    }
    __syncthreads();
    if (tid < H_SZ) {
        float acc = egb1[tid];
        for (int k = 0; k < 900; ++k) acc += egin[k] * egw1[k * H_SZ + tid];
        hid[tid] = acc > 0.f ? acc : 0.f;
    }
    __syncthreads();
    if (tid < D_SZ) {
        float acc = egb2[tid];
        for (int h = 0; h < H_SZ; ++h) acc += hid[h] * egw2[h * D_SZ + tid];
        eql[tid] = acc; eqf[tid] = acc; eqh[tid] = (_Float16)acc;
    }
    __syncthreads();
    if (tid < H_SZ) {
        float acc = vnb1[tid];
        for (int d = 0; d < D_SZ; ++d) acc += eql[d] * vnw1[d * H_SZ + tid];
        hid[tid] = acc > 0.f ? acc : 0.f;
    }
    __syncthreads();
    if (tid == 0) {
        float acc = vnb2[0];
        for (int h = 0; h < H_SZ; ++h) acc += hid[h] * vnw2[h];
        valueOut[0] = acc;
    }
}

// ---------------- assemble doc-scorer A (expanded-query cols + pad cols) ----------------
__global__ void k_assemble_dsA(const _Float16* __restrict__ eqh, _Float16* dsA) {
    int i = blockIdx.x * 256 + threadIdx.x;      // over NCPAD*308
    if (i >= NCPAD * 308) return;
    int c = i / 308, j = i - c * 308;
    int k = (j < D_SZ) ? j : (600 + (j - D_SZ)); // cols 0..299 and 600..607
    dsA[(size_t)c * KPAD + k] = (j < D_SZ) ? eqh[j] : (_Float16)0.f;
}

// ---------------- candidate embedding gather (bandwidth-dominant) ----------------
// writes doc-mean f16 into dsA cols 300..599
__global__ __launch_bounds__(320)
void k_doc_embed(const int* __restrict__ cand, const float* __restrict__ emb,
                 _Float16* dsA) {
    __shared__ int ids[512];
    int c = blockIdx.x, tid = threadIdx.x;
    _Float16* row = dsA + (size_t)c * KPAD + D_SZ;
    if (c >= NCAND) {
        if (tid < D_SZ) row[tid] = (_Float16)0.f;
        return;
    }
    for (int i = tid; i < 512; i += 320) ids[i] = cand[c * 512 + i];
    __syncthreads();
    if (tid < D_SZ) {
        float acc = 0.f;
        for (int t = 0; t < 512; ++t) {
            if (t + 8 < 512)
                __builtin_prefetch(&emb[(long)ids[t + 8] * D_SZ + tid], 0, 0);
            acc += emb[(long)ids[t] * D_SZ + tid];
        }
        row[tid] = (_Float16)(acc * (1.f / 512.f));
    }
}

// ---------------- doc scorer: [1008,608]f16 x [608,128]f16 via WMMA ----------------
// A tile is a pure linear copy from dsA -> LDS (async-to-LDS when available)
__global__ __launch_bounds__(128)
void k_doc_score(const _Float16* __restrict__ dsA,
                 const _Float16* __restrict__ w1p, const float* __restrict__ b1,
                 const float* __restrict__ w2, const float* __restrict__ b2v,
                 float* scores) {
    __shared__ __align__(16) _Float16 At[16 * KPAD];
    __shared__ float Hid[16 * H_SZ];
    __shared__ float Red[128];

    const int tid = threadIdx.x, lane = tid & 31, wave = tid >> 5;
    const int rowBase = blockIdx.x * 16;
    const _Float16* gsrc = dsA + (size_t)rowBase * KPAD;   // 16 rows contiguous

#if HAVE_ASYNC_LDS
    for (int i = tid; i < (16 * KPAD) / 8; i += 128) {
        __builtin_amdgcn_global_load_async_to_lds_b128(
            (GAS v4i*)(gsrc + i * 8), (LAS v4i*)(At + i * 8), 0, 0);
    }
#if __has_builtin(__builtin_amdgcn_s_wait_asynccnt)
    __builtin_amdgcn_s_wait_asynccnt(0);
#else
    asm volatile("s_wait_asynccnt 0x0" ::: "memory");
#endif
    __syncthreads();
#else
    for (int i = tid; i < (16 * KPAD) / 8; i += 128) {
        *(v8h*)(At + i * 8) = *(const v8h*)(gsrc + i * 8);
    }
    __syncthreads();
#endif

    for (int nt = wave; nt < 8; nt += 4) {
        int col = nt * 16 + (lane & 15);
        float bias = b1[col];
        v8f c;
#pragma unroll
        for (int r = 0; r < 8; ++r) c[r] = bias;
#pragma unroll
        for (int kc = 0; kc < NKC; ++kc) {
            v16h a = frag_a_from_lds(At, lane, kc);
            v16h b = frag_b_packed(w1p, nt, kc, lane);
            c = __builtin_amdgcn_wmma_f32_16x16x32_f16(
                    false, a, false, b, (short)0, c, false, false);
        }
        int mb = (lane & 16) ? 8 : 0;
#pragma unroll
        for (int r = 0; r < 8; ++r) {
            float h = c[r];
            Hid[(mb + r) * H_SZ + col] = h > 0.f ? h : 0.f;
        }
    }
    __syncthreads();

    {
        int r = tid >> 3, part = tid & 7;
        float acc = 0.f;
#pragma unroll
        for (int h = 0; h < 16; ++h)
            acc += Hid[r * H_SZ + part * 16 + h] * w2[part * 16 + h];
        Red[tid] = acc;
    }
    __syncthreads();
    if (tid < 16) {
        int c = rowBase + tid;
        if (c < NCAND) {
            float s = b2v[0];
#pragma unroll
            for (int p = 0; p < 8; ++p) s += Red[tid * 8 + p];
            scores[c] = s;
        }
    }
}

// ---------------- softmax over 1000 scores ----------------
__global__ __launch_bounds__(1024)
void k_softmax(const float* __restrict__ scores, float* out) {
    __shared__ float red[1024];
    int tid = threadIdx.x;
    float v = (tid < NCAND) ? scores[tid] : -3.4e38f;
    red[tid] = v;
    __syncthreads();
    for (int s = 512; s > 0; s >>= 1) {
        if (tid < s) red[tid] = fmaxf(red[tid], red[tid + s]);
        __syncthreads();
    }
    float mx = red[0];
    __syncthreads();
    float e = (tid < NCAND) ? __expf(v - mx) : 0.f;
    red[tid] = e;
    __syncthreads();
    for (int s = 512; s > 0; s >>= 1) {
        if (tid < s) red[tid] += red[tid + s];
        __syncthreads();
    }
    float inv = 1.f / red[0];
    if (tid < NCAND) out[tid] = e * inv;
}

__global__ void k_finalize(const float* __restrict__ valuep,
                           const int* __restrict__ topIds,
                           const float* __restrict__ topW, float* out) {
    int tid = threadIdx.x;
    if (tid == 0) out[1000] = valuep[0];
    if (tid < 10) {
        out[1001 + tid] = (float)topIds[tid];
        out[1011 + tid] = topW[tid];
    }
}

// ---------------- launcher ----------------
extern "C" void kernel_launch(void* const* d_in, const int* in_sizes, int n_in,
                              void* d_out, int out_size, void* d_ws, size_t ws_size,
                              hipStream_t stream) {
    const int*   query = (const int*)d_in[0];
    const int*   cand  = (const int*)d_in[1];
    const int*   fb    = (const int*)d_in[2];
    // d_in[3] = n_terms (==10, fixed output shape)
    const float* emb   = (const float*)d_in[4];
    const float* ts_w1 = (const float*)d_in[5];
    const float* ts_b1 = (const float*)d_in[6];
    const float* ts_w2 = (const float*)d_in[7];
    const float* ts_b2 = (const float*)d_in[8];
    const float* eg_w1 = (const float*)d_in[9];
    const float* eg_b1 = (const float*)d_in[10];
    const float* eg_w2 = (const float*)d_in[11];
    const float* eg_b2 = (const float*)d_in[12];
    const float* ds_w1 = (const float*)d_in[13];
    const float* ds_b1 = (const float*)d_in[14];
    const float* ds_w2 = (const float*)d_in[15];
    const float* ds_b2 = (const float*)d_in[16];
    const float* vn_w1 = (const float*)d_in[17];
    const float* vn_b1 = (const float*)d_in[18];
    const float* vn_w2 = (const float*)d_in[19];
    const float* vn_b2 = (const float*)d_in[20];

    char* ws = (char*)d_ws;
    float*    weights = (float*)(ws + OFF_WEIGHTS);
    float*    qctx    = (float*)(ws + OFF_QCTX);
    int*      topIds  = (int*)(ws + OFF_TOPIDS);
    float*    topW    = (float*)(ws + OFF_TOPW);
    float*    valuep  = (float*)(ws + OFF_VALUE);
    float*    eqf     = (float*)(ws + OFF_EQF);
    _Float16* eqh     = (_Float16*)(ws + OFF_EQH);
    _Float16* tsW1p   = (_Float16*)(ws + OFF_TSW1P);
    _Float16* dsW1p   = (_Float16*)(ws + OFF_DSW1P);
    _Float16* dsA     = (_Float16*)(ws + OFF_DSA);
    float*    scores  = (float*)(ws + OFF_SCORES);
    float*    out     = (float*)d_out;

    k_zero_weights<<<(V_SZ + 255) / 256, 256, 0, stream>>>(weights);
    k_pack_w1<<<(8 * NKC * 32 * 16 + 255) / 256, 256, 0, stream>>>(
        ts_w1, ds_w1, tsW1p, dsW1p);
    k_qctx<<<1, 320, 0, stream>>>(query, emb, qctx);
    k_term_score<<<NFB / 16, 128, 0, stream>>>(fb, emb, qctx, tsW1p, ts_b1,
                                               ts_w2, ts_b2, weights);
    k_topk<<<1, 256, 0, stream>>>(weights, topIds, topW);
    k_expand<<<1, 320, 0, stream>>>(topIds, emb, qctx,
                                    eg_w1, eg_b1, eg_w2, eg_b2,
                                    vn_w1, vn_b1, vn_w2, vn_b2,
                                    eqf, eqh, valuep);
    k_assemble_dsA<<<(NCPAD * 308 + 255) / 256, 256, 0, stream>>>(eqh, dsA);
    k_doc_embed<<<NCPAD, 320, 0, stream>>>(cand, emb, dsA);
    k_doc_score<<<NCPAD / 16, 128, 0, stream>>>(dsA, dsW1p, ds_b1,
                                                ds_w2, ds_b2, scores);
    k_softmax<<<1, 1024, 0, stream>>>(scores, out);
    k_finalize<<<1, 32, 0, stream>>>(valuep, topIds, topW, out);
}
